// LocalCapsulePooling_77309411328708
// MI455X (gfx1250) — matrix operand order repaired and checked by reading, not compiled, not run on hardware
//
#include <hip/hip_runtime.h>
#include <hip/hip_bf16.h>

// LocalCapsulePooling on gfx1250 (MI455X).
// GEMM x@W runs on v_wmma_f32_16x16x32_bf16 (wave32 WMMA) with ds_load_b128
// fragment loads from padded LDS; all segment ops are wave-per-edge float4
// gather + f32 atomic scatter (bandwidth-bound phase).

#define H 128
#define LDP (H + 8)          // padded LDS row stride: 272B = 4-bank shift/row
#define EPS_BN 1e-5f
#define EPS_SQ 1e-8f

typedef __attribute__((ext_vector_type(16))) __bf16 v16bf;
typedef __attribute__((ext_vector_type(8)))  __bf16 v8bf;
typedef __attribute__((ext_vector_type(8)))  float  v8f;

// ---------- helpers ----------
__device__ __forceinline__ unsigned f2key(float v) {
  unsigned u = __float_as_uint(v);
  return (u & 0x80000000u) ? ~u : (u | 0x80000000u); // monotone float->uint
}
__device__ __forceinline__ float key2f(unsigned k) {
  unsigned u = (k & 0x80000000u) ? (k ^ 0x80000000u) : ~k;
  return __uint_as_float(u);
}
__device__ __forceinline__ float wave_sum32(float v) {
#pragma unroll
  for (int m = 16; m > 0; m >>= 1) v += __shfl_xor(v, m, 32);
  return v;
}

// ---------- fill / init ----------
__global__ void fill_f32(float* p, int n, float v) {
  int i = blockIdx.x * blockDim.x + threadIdx.x;
  if (i < n) p[i] = v;
}
__global__ void rsqrt_inplace(float* p, int n) {
  int i = blockIdx.x * blockDim.x + threadIdx.x;
  if (i < n) p[i] = rsqrtf(p[i]);   // deg >= 1 always (self loops)
}

// ---------- edge preparation: int64 -> int32, self loops, degree ----------
__global__ void build_edges(const long long* __restrict__ ei, int E, int N,
                            int* __restrict__ row, int* __restrict__ col,
                            float* __restrict__ deg) {
  int i = blockIdx.x * blockDim.x + threadIdx.x;
  if (i < E) {
    int r = (int)ei[i];
    int c = (int)ei[(long long)E + i];
    row[i] = r; col[i] = c;
    atomicAdd(&deg[c], 1.0f);          // deg pre-filled with 1.0 (self loop)
  }
  if (i < N) { row[E + i] = i; col[E + i] = i; }
}

// ---------- GEMM: xw = x @ W via WMMA bf16 ----------
// Block = 256 thr (8 waves). Wave w handles rows [blk*128 + w*16, +16).
// LDS: W transposed as bf16 + 128-row x tile as bf16, rows padded to LDP.
// Fragment loads are contiguous 16B runs -> ds_load_b128.
__global__ __launch_bounds__(256) void gemm_xw_wmma(const float* __restrict__ x,
                                                    const float* __restrict__ W,
                                                    float* __restrict__ xw, int N) {
  __shared__ __bf16 sWt[H * LDP];   // sWt[n*LDP + k] = W[k][n]  (K-major)
  __shared__ __bf16 sX[128 * LDP];  // row-major x tile
  const int t = threadIdx.x;

  // W: coalesced float4 global read, transposed bf16 LDS stores
  for (int i = t * 4; i < H * H; i += 1024) {
    int k = i >> 7, n = i & 127;
    float4 w4 = *(const float4*)(W + i);
    sWt[(n + 0) * LDP + k] = (__bf16)w4.x;
    sWt[(n + 1) * LDP + k] = (__bf16)w4.y;
    sWt[(n + 2) * LDP + k] = (__bf16)w4.z;
    sWt[(n + 3) * LDP + k] = (__bf16)w4.w;
  }
  const int row0 = blockIdx.x * 128;
  for (int i = t * 4; i < 128 * H; i += 1024) {
    int rl = i >> 7, c = i & 127;
    int r = row0 + rl;
    float4 v = make_float4(0.f, 0.f, 0.f, 0.f);
    if (r < N) v = *(const float4*)(x + (size_t)r * H + c);
    __bf16* d = &sX[rl * LDP + c];
    d[0] = (__bf16)v.x; d[1] = (__bf16)v.y; d[2] = (__bf16)v.z; d[3] = (__bf16)v.w;
  }
  __syncthreads();

  const int wave = t >> 5, lane = t & 31;
  const int mtile = row0 + wave * 16;
  if (mtile >= N) return;
  const int half = lane >> 4, lid = lane & 15;

  v8f acc[8];
#pragma unroll
  for (int nt = 0; nt < 8; ++nt) acc[nt] = v8f{};

  union Frag { v16bf v; v8bf h[2]; };

  const __bf16* arow = &sX[(wave * 16 + lid) * LDP];
#pragma unroll
  for (int kk = 0; kk < H; kk += 32) {
    // A 16x32 bf16: lane(half,m): VGPR0-7 = K kk+16*half+[0..16)
    Frag a;
    a.h[0] = *(const v8bf*)(arow + kk + 16 * half);
    a.h[1] = *(const v8bf*)(arow + kk + 16 * half + 8);
#pragma unroll
    for (int nt = 0; nt < 8; ++nt) {
      // B 32x16 bf16: lane(half,n): VGPR j = K kk+16*half+{2j,2j+1}
      // -> 16 contiguous K values of column n = row n of sWt
      const __bf16* brow = &sWt[(nt * 16 + lid) * LDP];
      Frag b;
      b.h[0] = *(const v8bf*)(brow + kk + 16 * half);
      b.h[1] = *(const v8bf*)(brow + kk + 16 * half + 8);
      acc[nt] = __builtin_amdgcn_wmma_f32_16x16x32_bf16(
          false, a.v, false, b.v, (short)0, acc[nt], false, false);
    }
  }
  // C/D 16x16 f32 layout: lane n=l&15, VGPR r -> row r + 8*(l>>4)
  if (mtile + 16 <= N) {
#pragma unroll
    for (int nt = 0; nt < 8; ++nt)
#pragma unroll
      for (int r = 0; r < 8; ++r)
        xw[(size_t)(mtile + r + 8 * half) * H + nt * 16 + lid] = acc[nt][r];
  } else {
#pragma unroll
    for (int nt = 0; nt < 8; ++nt)
#pragma unroll
      for (int r = 0; r < 8; ++r) {
        int m = r + 8 * half;
        if (mtile + m < N)
          xw[(size_t)(mtile + m) * H + nt * 16 + lid] = acc[nt][r];
      }
  }
}

// ---------- GCN aggregate: x_pool[col] += dinv[row]*dinv[col] * xw[row] ----------
__global__ __launch_bounds__(256) void gcn_scatter(const int* __restrict__ row,
                                                   const int* __restrict__ col,
                                                   const float* __restrict__ dinv,
                                                   const float* __restrict__ xw,
                                                   float* __restrict__ xp, int Etot) {
  int w = (blockIdx.x * blockDim.x + threadIdx.x) >> 5;
  int lane = threadIdx.x & 31;
  if (w >= Etot) return;
  int r = row[w], c = col[w];
  float nrm = dinv[r] * dinv[c];
  float4 v = ((const float4*)(xw + (size_t)r * H))[lane];
  float* dst = xp + (size_t)c * H + lane * 4;
  atomicAdd(dst + 0, nrm * v.x);
  atomicAdd(dst + 1, nrm * v.y);
  atomicAdd(dst + 2, nrm * v.z);
  atomicAdd(dst + 3, nrm * v.w);
}

// ---------- batch-norm stats: one block per feature ----------
__global__ __launch_bounds__(256) void bn_stats(const float* __restrict__ z, int N,
                                                float* __restrict__ mu,
                                                float* __restrict__ var) {
  __shared__ float sh[256], sh2[256];
  int f = blockIdx.x;
  float s = 0.f, s2 = 0.f;
  for (int r = threadIdx.x; r < N; r += 256) {
    float v = z[(size_t)r * H + f];
    s += v; s2 += v * v;
  }
  sh[threadIdx.x] = s; sh2[threadIdx.x] = s2;
  __syncthreads();
  for (int o = 128; o > 0; o >>= 1) {
    if (threadIdx.x < o) { sh[threadIdx.x] += sh[threadIdx.x + o]; sh2[threadIdx.x] += sh2[threadIdx.x + o]; }
    __syncthreads();
  }
  if (threadIdx.x == 0) {
    float m = sh[0] / (float)N;
    mu[f] = m;
    var[f] = sh2[0] / (float)N - m * m;
  }
}

// ---------- (optional BN) + squash, one wave per node ----------
__global__ __launch_bounds__(256) void bn_squash(float* __restrict__ z, int N,
                                                 const float* __restrict__ mu,
                                                 const float* __restrict__ var,
                                                 const float* __restrict__ gamma,
                                                 const float* __restrict__ beta,
                                                 int do_bn) {
  int w = (blockIdx.x * blockDim.x + threadIdx.x) >> 5;
  int lane = threadIdx.x & 31;
  if (w >= N) return;
  float4 v = ((const float4*)(z + (size_t)w * H))[lane];
  if (do_bn) {
    int f = lane * 4;
    v.x = gamma[f + 0] * (v.x - mu[f + 0]) * rsqrtf(var[f + 0] + EPS_BN) + beta[f + 0];
    v.y = gamma[f + 1] * (v.y - mu[f + 1]) * rsqrtf(var[f + 1] + EPS_BN) + beta[f + 1];
    v.z = gamma[f + 2] * (v.z - mu[f + 2]) * rsqrtf(var[f + 2] + EPS_BN) + beta[f + 2];
    v.w = gamma[f + 3] * (v.w - mu[f + 3]) * rsqrtf(var[f + 3] + EPS_BN) + beta[f + 3];
  }
  float n2 = wave_sum32(v.x * v.x + v.y * v.y + v.z * v.z + v.w * v.w);
  float s = (n2 / (1.0f + n2)) * rsqrtf(n2 + EPS_SQ);
  v.x *= s; v.y *= s; v.z *= s; v.w *= s;
  ((float4*)(z + (size_t)w * H))[lane] = v;
}

// ---------- segment softmax over col ----------
__global__ void seg_max(const float* __restrict__ bij, const int* __restrict__ col,
                        unsigned* __restrict__ smax, int Etot) {
  int e = blockIdx.x * blockDim.x + threadIdx.x;
  if (e < Etot) atomicMax(&smax[col[e]], f2key(bij[e]));
}
__global__ void seg_expsum(const float* __restrict__ bij, const int* __restrict__ col,
                           const unsigned* __restrict__ smax, float* __restrict__ ev,
                           float* __restrict__ ssum, int Etot) {
  int e = blockIdx.x * blockDim.x + threadIdx.x;
  if (e >= Etot) return;
  float m = key2f(smax[col[e]]);
  float x = __expf(bij[e] - m);
  ev[e] = x;
  atomicAdd(&ssum[col[e]], x);
}
__global__ void seg_norm(const float* __restrict__ ev, const int* __restrict__ col,
                         const float* __restrict__ ssum, float* __restrict__ cij, int Etot) {
  int e = blockIdx.x * blockDim.x + threadIdx.x;
  if (e < Etot) cij[e] = ev[e] / (ssum[col[e]] + 1e-16f);
}

// ---------- routing scatter: cr[row] += c_ij * x_pool[col] ----------
__global__ __launch_bounds__(256) void route_scatter(const int* __restrict__ row,
                                                     const int* __restrict__ col,
                                                     const float* __restrict__ cij,
                                                     const float* __restrict__ xp,
                                                     float* __restrict__ cr, int Etot) {
  int w = (blockIdx.x * blockDim.x + threadIdx.x) >> 5;
  int lane = threadIdx.x & 31;
  if (w >= Etot) return;
  float c = cij[w];
  int r = row[w], cc = col[w];
  float4 v = ((const float4*)(xp + (size_t)cc * H))[lane];
  float* dst = cr + (size_t)r * H + lane * 4;
  atomicAdd(dst + 0, c * v.x);
  atomicAdd(dst + 1, c * v.y);
  atomicAdd(dst + 2, c * v.z);
  atomicAdd(dst + 3, c * v.w);
}

// ---------- b_ij += dot(cr[row], x_pool[col]) ----------
__global__ __launch_bounds__(256) void dot_update(const int* __restrict__ row,
                                                  const int* __restrict__ col,
                                                  const float* __restrict__ cr,
                                                  const float* __restrict__ xp,
                                                  float* __restrict__ bij, int Etot) {
  int w = (blockIdx.x * blockDim.x + threadIdx.x) >> 5;
  int lane = threadIdx.x & 31;
  if (w >= Etot) return;
  float4 a = ((const float4*)(cr + (size_t)row[w] * H))[lane];
  float4 b = ((const float4*)(xp + (size_t)col[w] * H))[lane];
  float d = wave_sum32(a.x * b.x + a.y * b.y + a.z * b.z + a.w * b.w);
  if (lane == 0) bij[w] += d;
}

// ---------- score keys: key(||cr_i||) ----------
__global__ __launch_bounds__(256) void score_keys(const float* __restrict__ cr, int N,
                                                  unsigned* __restrict__ keys) {
  int w = (blockIdx.x * blockDim.x + threadIdx.x) >> 5;
  int lane = threadIdx.x & 31;
  if (w >= N) return;
  float4 v = ((const float4*)(cr + (size_t)w * H))[lane];
  float n2 = wave_sum32(v.x * v.x + v.y * v.y + v.z * v.z + v.w * v.w);
  if (lane == 0) keys[w] = f2key(sqrtf(n2));
}

// ---------- exact top-k rank (descending, stable by lower index) ----------
__global__ void rank_topk(const unsigned* __restrict__ keys, int N,
                          int* __restrict__ rank) {
  int i = blockIdx.x * blockDim.x + threadIdx.x;
  if (i >= N) return;
  unsigned ki = keys[i];
  int cnt = 0;
  for (int j = 0; j < N; ++j) {            // keys (200KB) are L2/L1 resident
    unsigned kj = keys[j];
    cnt += (kj > ki) || (kj == ki && j < i);
  }
  rank[i] = cnt;
}

// ---------- outputs: x_out rows + perm ----------
__global__ __launch_bounds__(256) void gather_out(const float* __restrict__ cr,
                                                  const int* __restrict__ rank, int N,
                                                  int K, float* __restrict__ x_out,
                                                  float* __restrict__ perm) {
  int w = (blockIdx.x * blockDim.x + threadIdx.x) >> 5;
  int lane = threadIdx.x & 31;
  if (w >= N) return;
  int rk = rank[w];
  if (rk >= K) return;
  float4 v = ((const float4*)(cr + (size_t)w * H))[lane];
  ((float4*)(x_out + (size_t)rk * H))[lane] = v;
  if (lane == 0) perm[rk] = (float)w;
}

// ---------- S_value: kept[row] ? c_ij : 0 ----------
__global__ void write_sval(const int* __restrict__ row, const int* __restrict__ rank,
                           const float* __restrict__ cij, int K,
                           float* __restrict__ sval, int Etot) {
  int e = blockIdx.x * blockDim.x + threadIdx.x;
  if (e < Etot) sval[e] = (rank[row[e]] < K) ? cij[e] : 0.0f;
}

// ================================================================
extern "C" void kernel_launch(void* const* d_in, const int* in_sizes, int n_in,
                              void* d_out, int out_size, void* d_ws, size_t ws_size,
                              hipStream_t stream) {
  const float*     x      = (const float*)d_in[0];
  const long long* ei     = (const long long*)d_in[1];   // int64 edge_index [2,E]
  const float*     W      = (const float*)d_in[2];
  // d_in[3] = b: cancels exactly under batch_norm (mean shift) -> skipped
  const float*     gamma1 = (const float*)d_in[4];
  const float*     beta1  = (const float*)d_in[5];
  const float*     gamma2 = (const float*)d_in[6];
  const float*     beta2  = (const float*)d_in[7];

  const int N = in_sizes[0] / H;
  const int E = in_sizes[1] / 2;
  const int Etot = E + N;
  const int K = N / 2;

  // bump-allocate workspace (256B aligned)
  char* wsp = (char*)d_ws;
  size_t off = 0;
  auto alloc = [&](size_t bytes) -> char* {
    char* p = wsp + off;
    off = (off + bytes + 255) & ~(size_t)255;
    return p;
  };
  float*    xw   = (float*)alloc((size_t)N * H * 4);
  float*    xp   = (float*)alloc((size_t)N * H * 4);   // x_pool
  float*    cr   = (float*)alloc((size_t)N * H * 4);
  float*    dinv = (float*)alloc((size_t)N * 4);       // deg -> rsqrt(deg)
  unsigned* smax = (unsigned*)alloc((size_t)N * 4);
  float*    ssum = (float*)alloc((size_t)N * 4);
  unsigned* keys = (unsigned*)alloc((size_t)N * 4);
  int*      rank = (int*)alloc((size_t)N * 4);
  float*    mu   = (float*)alloc(H * 4);
  float*    var  = (float*)alloc(H * 4);
  int*      row  = (int*)alloc((size_t)Etot * 4);
  int*      col  = (int*)alloc((size_t)Etot * 4);
  float*    bij  = (float*)alloc((size_t)Etot * 4);
  float*    ev   = (float*)alloc((size_t)Etot * 4);
  float*    cij  = (float*)alloc((size_t)Etot * 4);

  float* x_out = (float*)d_out;                 // [K, H]
  float* perm  = x_out + (size_t)K * H;         // [K] (indices as floats)
  float* sval  = perm + K;                      // [Etot]

  const int T = 256;
  const int gN  = (N + T - 1) / T;
  const int gE  = (Etot + T - 1) / T;
  const int gEE = (((E > N) ? E : N) + T - 1) / T;
  const int gNw = (N + 7) / 8;                  // wave-per-node
  const int gEw = (Etot + 7) / 8;               // wave-per-edge

  // --- GCN conv ---
  fill_f32<<<gN, T, 0, stream>>>(dinv, N, 1.0f);                       // self-loop deg
  build_edges<<<gEE, T, 0, stream>>>(ei, E, N, row, col, dinv);
  rsqrt_inplace<<<gN, T, 0, stream>>>(dinv, N);
  gemm_xw_wmma<<<(N + 127) / 128, 256, 0, stream>>>(x, W, xw, N);      // WMMA
  hipMemsetAsync(xp, 0, (size_t)N * H * 4, stream);
  gcn_scatter<<<gEw, T, 0, stream>>>(row, col, dinv, xw, xp, Etot);
  bn_stats<<<H, T, 0, stream>>>(xp, N, mu, var);
  bn_squash<<<gNw, T, 0, stream>>>(xp, N, mu, var, gamma1, beta1, 1);

  // --- dynamic routing (3 iters; first 2 "detached" = numerically identical) ---
  fill_f32<<<gE, T, 0, stream>>>(bij, Etot, 1.0f);
  for (int it = 0; it < 3; ++it) {
    hipMemsetAsync(smax, 0, (size_t)N * 4, stream);   // key(0) < any real key
    hipMemsetAsync(ssum, 0, (size_t)N * 4, stream);
    seg_max<<<gE, T, 0, stream>>>(bij, col, smax, Etot);
    seg_expsum<<<gE, T, 0, stream>>>(bij, col, smax, ev, ssum, Etot);
    seg_norm<<<gE, T, 0, stream>>>(ev, col, ssum, cij, Etot);
    hipMemsetAsync(cr, 0, (size_t)N * H * 4, stream);
    route_scatter<<<gEw, T, 0, stream>>>(row, col, cij, xp, cr, Etot);
    if (it < 2) {
      bn_squash<<<gNw, T, 0, stream>>>(cr, N, mu, var, (const float*)nullptr,
                                       (const float*)nullptr, 0);      // squash only
      dot_update<<<gEw, T, 0, stream>>>(row, col, cr, xp, bij, Etot);
    } else {
      bn_stats<<<H, T, 0, stream>>>(cr, N, mu, var);
      bn_squash<<<gNw, T, 0, stream>>>(cr, N, mu, var, gamma2, beta2, 1);
    }
  }

  // --- top-k pooling + outputs ---
  score_keys<<<gNw, T, 0, stream>>>(cr, N, keys);
  rank_topk<<<gN, T, 0, stream>>>(keys, N, rank);
  gather_out<<<gNw, T, 0, stream>>>(cr, rank, N, K, x_out, perm);
  write_sval<<<gE, T, 0, stream>>>(row, rank, cij, K, sval, Etot);
}
